// GGAT_77360950936033
// MI455X (gfx1250) — compile-verified, matrix operand order checked by reading, not run on hardware
//
#include <hip/hip_runtime.h>
#include <hip/hip_bf16.h>

#define N_NODES 50000
#define N_EDGES 800000
#define IN_FEAT 256
#define HEADS   8
#define D_K     64
#define OUTF    (HEADS * D_K)   // 512
#define NTILES  (OUTF / 16)     // 32
#define KTILES  (IN_FEAT / 32)  // 8

typedef __attribute__((ext_vector_type(8)))  __bf16 v8bf;
typedef __attribute__((ext_vector_type(16))) __bf16 v16bf;
typedef __attribute__((ext_vector_type(8)))  float  v8f;

__device__ __forceinline__ __bf16 f2bf(float f) {
    unsigned u = __float_as_uint(f);
    unsigned r = u + 0x7FFFu + ((u >> 16) & 1u);      // round-to-nearest-even
    unsigned short h = (unsigned short)(r >> 16);
    return __builtin_bit_cast(__bf16, h);
}

// order-preserving float -> uint encoding for atomicMax-based segment max
__device__ __forceinline__ unsigned encf(float f) {
    unsigned u = __float_as_uint(f);
    return (u & 0x80000000u) ? ~u : (u | 0x80000000u);
}
__device__ __forceinline__ float decf(unsigned u) {
    u = (u & 0x80000000u) ? (u & 0x7FFFFFFFu) : ~u;
    return __uint_as_float(u);
}

// ---------------------------------------------------------------------------
// 0) zero-init out / smax / denom (harness poisons buffers with 0xAA)
// ---------------------------------------------------------------------------
__global__ void ggat_init(float* __restrict__ out, unsigned* __restrict__ smax,
                          float* __restrict__ denom) {
    int i = blockIdx.x * blockDim.x + threadIdx.x;
    if (i < N_NODES * OUTF) out[i] = 0.0f;
    if (i < N_NODES * HEADS) { smax[i] = 0u; denom[i] = 0.0f; }
}

// ---------------------------------------------------------------------------
// 0b) convert x (f32 row-major) -> xb (bf16 row-major), 4 elems/thread
// ---------------------------------------------------------------------------
__global__ void __launch_bounds__(256)
cvt_x(const float* __restrict__ x, __bf16* __restrict__ xb) {
    int i = (blockIdx.x * blockDim.x + threadIdx.x) * 4;
    if (i >= N_NODES * IN_FEAT) return;
    #pragma unroll
    for (int j = 0; j < 4; ++j) xb[i + j] = f2bf(x[i + j]);
}

// ---------------------------------------------------------------------------
// 0c) pre-swizzle Wq/Wk/Wv into per-lane WMMA B-fragment order (bf16).
//     Layout: Wp[ ((which*32 + nt)*8 + kt)*512 + lane*16 + j ]
//     j -> K = kt*32 + (j&7) + (lane>>4)*8 + (j>>3)*16 ; N = nt*16 + (lane&15)
// ---------------------------------------------------------------------------
__global__ void __launch_bounds__(256)
pack_w(const float* __restrict__ Wq, const float* __restrict__ Wk,
       const float* __restrict__ Wv, __bf16* __restrict__ Wp) {
    int t = blockIdx.x * blockDim.x + threadIdx.x;       // 0 .. 3*32*8*32-1
    if (t >= 3 * NTILES * KTILES * 32) return;
    const int lane  = t & 31;
    const int kt    = (t >> 5) & 7;
    const int nt    = (t >> 8) & 31;
    const int which = t >> 13;
    const float* __restrict__ W = (which == 0) ? Wq : (which == 1) ? Wk : Wv;
    const int kh = (lane >> 4) * 8;
    const int n  = nt * 16 + (lane & 15);
    __bf16* __restrict__ dst = Wp + (size_t)t * 16;
    #pragma unroll
    for (int j = 0; j < 16; ++j) {
        int k = kt * 32 + (j & 7) + kh + ((j >> 3) * 16);
        dst[j] = f2bf(W[(size_t)k * OUTF + n]);
    }
}

// ---------------------------------------------------------------------------
// 1) QKV GEMM: C[50000,512] = xb[50000,256] @ W[256,512], bf16 WMMA, f32 acc.
//    grid = (3125 M-tiles, 3 matrices), block = 128 (4 waves).
//    Per kt: 2 A b128 loads, then ALL 8 B fragment loads issued up-front so
//    WMMA t can retire while loads for t+1..7 are still in flight.
// ---------------------------------------------------------------------------
__global__ void __launch_bounds__(128)
qkv_gemm(const __bf16* __restrict__ xb, const __bf16* __restrict__ Wp,
         float* __restrict__ Q, float* __restrict__ K, float* __restrict__ V) {
    const int mtile = blockIdx.x;          // 0..3124
    const int which = blockIdx.y;          // 0=Q 1=K 2=V
    float* __restrict__ Out = (which == 0) ? Q : (which == 1) ? K : V;

    const int wave = threadIdx.x >> 5;     // 0..3
    const int lane = threadIdx.x & 31;
    const int m0   = mtile * 16;
    const int mrow = m0 + (lane & 15);     // A row for this lane
    const int kh   = (lane >> 4) * 8;      // K sub-offset from lane half

    const __bf16* __restrict__ xrow = xb + (size_t)mrow * IN_FEAT + kh;
    // per-lane B fragment base for this wave's first N-tile
    const __bf16* __restrict__ wfrag =
        Wp + (((size_t)which * NTILES + wave * 8) * KTILES) * 512 + lane * 16;

    v8f acc[8];
    #pragma unroll
    for (int t = 0; t < 8; ++t) acc[t] = (v8f){};

    #pragma unroll 1
    for (int kt = 0; kt < KTILES; ++kt) {
        // A fragment: two contiguous 16B halves (K [kh,kh+8) and [kh+16,kh+24))
        v8bf alo = *(const v8bf*)(xrow + kt * 32);
        v8bf ahi = *(const v8bf*)(xrow + kt * 32 + 16);
        v16bf a = __builtin_shufflevector(alo, ahi,
                    0, 1, 2, 3, 4, 5, 6, 7, 8, 9, 10, 11, 12, 13, 14, 15);

        // issue all 8 B fragment loads (contiguous, 32B-aligned, coalesced)
        v16bf b[8];
        #pragma unroll
        for (int t = 0; t < 8; ++t)
            b[t] = *(const v16bf*)(wfrag + ((size_t)t * KTILES + kt) * 512);

        if (kt + 1 < KTILES) {
            __builtin_prefetch(xrow + (kt + 1) * 32, 0, 1);
            __builtin_prefetch(wfrag + (size_t)(kt + 1) * 512, 0, 1);
        }

        #pragma unroll
        for (int t = 0; t < 8; ++t)
            acc[t] = __builtin_amdgcn_wmma_f32_16x16x32_bf16(
                false, a, false, b[t], (short)0, acc[t], false, false);
    }

    // C layout: VGPR r, lanes 0-15 -> M=r, lanes 16-31 -> M=8+r
    const int mbase = m0 + 8 * (lane >> 4);
    #pragma unroll
    for (int t = 0; t < 8; ++t) {
        const int n = (wave * 8 + t) * 16 + (lane & 15);
        #pragma unroll
        for (int r = 0; r < 8; ++r)
            Out[(size_t)(mbase + r) * OUTF + n] = acc[t][r];
    }
}

// ---------------------------------------------------------------------------
// 2) Edge scores: wave per edge; lane covers 16 feats, head = lane>>2.
//    s_edge = sum_d (K-Q)^2 * D ; s_self = sum_d K*Q*D ; leaky_relu(s/8, 0.2)
// ---------------------------------------------------------------------------
__global__ void __launch_bounds__(256)
edge_scores(const float* __restrict__ Q, const float* __restrict__ K,
            const int* __restrict__ ei, const float* __restrict__ D,
            float* __restrict__ scores, unsigned* __restrict__ smax) {
    const int e = blockIdx.x * (blockDim.x >> 5) + (threadIdx.x >> 5);
    if (e >= N_EDGES) return;
    const int lane = threadIdx.x & 31;
    const int r = ei[e];
    const int c = ei[N_EDGES + e];
    const bool self = (r == c);

    const float* __restrict__ Kr = K + (size_t)r * OUTF + lane * 16;
    const float* __restrict__ Qc = Q + (size_t)c * OUTF + lane * 16;
    const float* __restrict__ Dl = D + lane * 16;  // flat [8,64] matches layout
    float partial = 0.0f;
    #pragma unroll
    for (int i = 0; i < 16; i += 4) {
        float4 kv = *(const float4*)(Kr + i);
        float4 qv = *(const float4*)(Qc + i);
        float4 dv = *(const float4*)(Dl + i);
        if (self) {
            partial += kv.x * qv.x * dv.x + kv.y * qv.y * dv.y +
                       kv.z * qv.z * dv.z + kv.w * qv.w * dv.w;
        } else {
            float dx = kv.x - qv.x, dy = kv.y - qv.y,
                  dz = kv.z - qv.z, dw = kv.w - qv.w;
            partial += dx * dx * dv.x + dy * dy * dv.y +
                       dz * dz * dv.z + dw * dw * dv.w;
        }
    }
    // reduce across the 4 lanes of each head
    partial += __shfl_xor(partial, 1, 32);
    partial += __shfl_xor(partial, 2, 32);

    if ((lane & 3) == 0) {
        const int h = lane >> 2;
        float s = partial * 0.125f;                    // 1/sqrt(64)
        s = (s > 0.0f) ? s : 0.2f * s;                 // leaky relu
        scores[(size_t)e * HEADS + h] = s;
        atomicMax(&smax[r * HEADS + h], encf(s));
    }
}

// ---------------------------------------------------------------------------
// 3) ex = exp(s - smax[row]); denom[row] += ex  (thread per edge*head)
// ---------------------------------------------------------------------------
__global__ void __launch_bounds__(256)
edge_exp(const int* __restrict__ ei, const unsigned* __restrict__ smax,
         float* __restrict__ scores, float* __restrict__ denom) {
    const int idx = blockIdx.x * blockDim.x + threadIdx.x;
    if (idx >= N_EDGES * HEADS) return;
    const int e = idx >> 3;
    const int h = idx & 7;
    const int r = ei[e];
    const float m  = decf(smax[r * HEADS + h]);
    const float ex = __expf(scores[idx] - m);
    scores[idx] = ex;                                  // reuse buffer for ex
    atomicAdd(&denom[r * HEADS + h], ex);
}

// ---------------------------------------------------------------------------
// 4) out[row] += (ex/denom[row]) * V[col]  (wave per edge, 16 feats/lane)
// ---------------------------------------------------------------------------
__global__ void __launch_bounds__(256)
aggregate(const float* __restrict__ V, const int* __restrict__ ei,
          const float* __restrict__ ex, const float* __restrict__ denom,
          float* __restrict__ out) {
    const int e = blockIdx.x * (blockDim.x >> 5) + (threadIdx.x >> 5);
    if (e >= N_EDGES) return;
    const int lane = threadIdx.x & 31;
    const int r = ei[e];
    const int c = ei[N_EDGES + e];
    const int h = lane >> 2;
    const float alpha = ex[(size_t)e * HEADS + h] / (denom[r * HEADS + h] + 1e-16f);

    const float* __restrict__ Vc = V + (size_t)c * OUTF + lane * 16;
    float* __restrict__       Or = out + (size_t)r * OUTF + lane * 16;
    #pragma unroll
    for (int i = 0; i < 16; i += 4) {
        float4 v = *(const float4*)(Vc + i);
        atomicAdd(&Or[i + 0], alpha * v.x);
        atomicAdd(&Or[i + 1], alpha * v.y);
        atomicAdd(&Or[i + 2], alpha * v.z);
        atomicAdd(&Or[i + 3], alpha * v.w);
    }
}

// ---------------------------------------------------------------------------
extern "C" void kernel_launch(void* const* d_in, const int* in_sizes, int n_in,
                              void* d_out, int out_size, void* d_ws, size_t ws_size,
                              hipStream_t stream) {
    const float* x  = (const float*)d_in[0];
    const int*   ei = (const int*)d_in[1];
    const float* Wq = (const float*)d_in[2];
    const float* Wk = (const float*)d_in[3];
    const float* Wv = (const float*)d_in[4];
    const float* D  = (const float*)d_in[5];
    float* out = (float*)d_out;

    // workspace layout: Q | K | V | scores/ex | smax | denom | xb | Wp
    const size_t NM = (size_t)N_NODES * OUTF;          // 25.6M floats
    float*    Q      = (float*)d_ws;
    float*    K      = Q + NM;
    float*    V      = K + NM;
    float*    scores = V + NM;                          // E*H = 6.4M
    unsigned* smax   = (unsigned*)(scores + (size_t)N_EDGES * HEADS);
    float*    denom  = (float*)(smax + (size_t)N_NODES * HEADS);
    __bf16*   xb     = (__bf16*)(denom + (size_t)N_NODES * HEADS);
    __bf16*   Wp     = xb + (size_t)N_NODES * IN_FEAT;  // 3*256*512 bf16

    // 0) init (covers 25.6M elements)
    ggat_init<<<(N_NODES * OUTF + 255) / 256, 256, 0, stream>>>(out, smax, denom);
    // 0b) x -> bf16
    cvt_x<<<(N_NODES * IN_FEAT / 4 + 255) / 256, 256, 0, stream>>>(x, xb);
    // 0c) pack W fragments
    pack_w<<<(3 * NTILES * KTILES * 32 + 255) / 256, 256, 0, stream>>>(Wq, Wk, Wv, Wp);
    // 1) QKV GEMM: 3125 M-tiles x 3 matrices, 128 threads
    {
        dim3 grid(N_NODES / 16, 3);
        qkv_gemm<<<grid, 128, 0, stream>>>(xb, Wp, Q, K, V);
    }
    // 2) edge scores: wave per edge, 8 waves/block
    edge_scores<<<N_EDGES / 8, 256, 0, stream>>>(Q, K, ei, D, scores, smax);
    // 3) exp + denom: thread per (edge, head)
    edge_exp<<<(N_EDGES * HEADS + 255) / 256, 256, 0, stream>>>(ei, smax, scores, denom);
    // 4) aggregation: wave per edge
    aggregate<<<N_EDGES / 8, 256, 0, stream>>>(V, ei, scores, denom, out);
}
